// E90GNN_74474732913082
// MI455X (gfx1250) — compile-verified, matrix operand order checked by reading, not run on hardware
//
#include <hip/hip_runtime.h>

#define NN 50000     // nodes
#define NE 800000    // edges
#define DIM 128      // feature dim
#define NL 3         // layers
#define NG 512       // graphs

typedef __attribute__((ext_vector_type(16))) __bf16 v16bf;
typedef __attribute__((ext_vector_type(8)))  __bf16 v8bf;
typedef __attribute__((ext_vector_type(8)))  float  v8f;

// ---------------------------------------------------------------- utilities
__global__ void fill_f32(float* __restrict__ p, float v, int n) {
  int i = blockIdx.x * blockDim.x + threadIdx.x;
  if (i < n) p[i] = v;
}

__global__ void deg_scatter(const int* __restrict__ col, float* __restrict__ deg, int e) {
  int i = blockIdx.x * blockDim.x + threadIdx.x;
  if (i < e) atomicAdd(&deg[col[i]], 1.0f);
}

__global__ void rsqrt_inplace(float* __restrict__ d, int n) {
  int i = blockIdx.x * blockDim.x + threadIdx.x;
  if (i < n) d[i] = rsqrtf(d[i]);   // deg >= 1 always (self loop)
}

__global__ void cvt_f32_bf16(const float* __restrict__ s, __bf16* __restrict__ d, int n) {
  int i = blockIdx.x * blockDim.x + threadIdx.x;
  if (i < n) d[i] = (__bf16)s[i];
}

// -------------------------------------------------- h = x @ W^T  via WMMA
// Block: 256 threads = 8 waves; block covers 128 rows x all 128 cols.
// W (32KB bf16) is staged once per block into LDS; each wave hoists its four
// A fragments (16 rows x full K=128) into registers with one load clause,
// then runs 8 column tiles x 4 K-steps of v_wmma_f32_16x16x32_bf16 entirely
// against ds_load_b128 B-fragment reads (low latency, no global stalls).
// A (16x32 bf16) lane layout: lane<16 -> M=lane, K in {k0..k0+7, k0+16..k0+23};
// lane>=16 -> M=lane-16, K in {k0+8..k0+15, k0+24..k0+31}. B = W^T so the B
// fragment for column n is two contiguous 8-elt runs of W row n (row-major).
__global__ void __launch_bounds__(256)
gemm_xwT_wmma(const __bf16* __restrict__ xb, const __bf16* __restrict__ wb,
              float* __restrict__ h, int nrows) {
  __shared__ __bf16 wlds[DIM * DIM];   // 32 KB

  const int lane = threadIdx.x & 31;
  const int wave = threadIdx.x >> 5;
  const int m0   = blockIdx.x * 128 + wave * 16;
  const int mr   = lane & 15;        // row within tile / col within tile
  const int kh   = lane >> 4;        // K-half select
  int arow = m0 + mr;
  if (arow >= nrows) arow = nrows - 1;   // clamp loads; stores guarded below

  // cooperative stage of W into LDS: 2048 x b128 moves, 8 per thread
  {
    const v8bf* __restrict__ src = (const v8bf*)wb;
    v8bf* dst = (v8bf*)wlds;
#pragma unroll
    for (int i = 0; i < 8; ++i)
      dst[threadIdx.x + i * 256] = src[threadIdx.x + i * 256];
  }
  __syncthreads();

  // hoist all four A fragments (one clause of 8 global_load_b128)
  v16bf a[4];
#pragma unroll
  for (int ks = 0; ks < 4; ++ks) {
    const __bf16* ap = xb + (size_t)arow * DIM + ks * 32 + kh * 8;
    v8bf alo = *(const v8bf*)(ap);
    v8bf ahi = *(const v8bf*)(ap + 16);
    a[ks] = __builtin_shufflevector(alo, ahi, 0,1,2,3,4,5,6,7,8,9,10,11,12,13,14,15);
  }

  const bool full = (m0 + 16 <= nrows);   // true for every block but the tail

#pragma unroll
  for (int ct = 0; ct < 8; ++ct) {
    const __bf16* bp = wlds + (ct * 16 + mr) * DIM + kh * 8;  // LDS, imm offsets per ks
    v8f acc = {};
#pragma unroll
    for (int ks = 0; ks < 4; ++ks) {
      v8bf blo = *(const v8bf*)(bp + ks * 32);
      v8bf bhi = *(const v8bf*)(bp + ks * 32 + 16);
      v16bf b = __builtin_shufflevector(blo, bhi, 0,1,2,3,4,5,6,7,8,9,10,11,12,13,14,15);
      acc = __builtin_amdgcn_wmma_f32_16x16x32_bf16(
          false, a[ks], false, b, (short)0, acc, false, false);
    }
    // C/D layout: lane 0-15 -> M = r, lane 16-31 -> M = 8 + r; N = lane & 15
    float* hp = h + (size_t)(m0 + kh * 8) * DIM + ct * 16 + mr;
    if (full) {
#pragma unroll
      for (int r = 0; r < 8; ++r) hp[(size_t)r * DIM] = acc[r];
    } else {
#pragma unroll
      for (int r = 0; r < 8; ++r)
        if (m0 + kh * 8 + r < nrows) hp[(size_t)r * DIM] = acc[r];
    }
  }
}

// ------------------------------------------ edge gather + atomic scatter-add
// One wave per edge; each lane owns 4 consecutive features (32*4 = 128).
__global__ void __launch_bounds__(256)
edge_scatter(const int* __restrict__ rowi, const int* __restrict__ coli,
             const float* __restrict__ dis, const float* __restrict__ h,
             float* __restrict__ agg, int e) {
  int gw   = (blockIdx.x * blockDim.x + threadIdx.x) >> 5;
  int lane = threadIdx.x & 31;
  if (gw >= e) return;
  int r = rowi[gw], c = coli[gw];
  float nrm = dis[r] * dis[c];
  const float4 hv = *(const float4*)(h + (size_t)r * DIM + lane * 4);
  float* dst = agg + (size_t)c * DIM + lane * 4;
  atomicAdd(dst + 0, hv.x * nrm);
  atomicAdd(dst + 1, hv.y * nrm);
  atomicAdd(dst + 2, hv.z * nrm);
  atomicAdd(dst + 3, hv.w * nrm);
}

// ----------------- self-loop contribution + bias + relu; emit f32 and bf16
__global__ void layer_finalize(float* __restrict__ agg, const float* __restrict__ h,
                               const float* __restrict__ dis, const float* __restrict__ bias,
                               __bf16* __restrict__ xb, int nelem) {
  int i = blockIdx.x * blockDim.x + threadIdx.x;
  if (i >= nelem) return;
  int node = i >> 7;
  int f    = i & 127;
  float s = dis[node];
  float v = agg[i] + h[i] * (s * s) + bias[f];
  v = fmaxf(v, 0.0f);
  agg[i] = v;             // f32 copy (used by pooling after last layer)
  xb[i]  = (__bf16)v;     // bf16 copy (next layer's WMMA A operand)
}

// ------------------------------------------------------------ mean pool
__global__ void pool_scatter(const float* __restrict__ x, const int* __restrict__ batch,
                             float* __restrict__ pool, float* __restrict__ cnt, int nelem) {
  int i = blockIdx.x * blockDim.x + threadIdx.x;
  if (i >= nelem) return;
  int node = i >> 7;
  int f    = i & 127;
  int g = batch[node];
  atomicAdd(&pool[g * DIM + f], x[i]);
  if (f == 0) atomicAdd(&cnt[g], 1.0f);
}

// one wave per graph: dot(pool[g], lin_w) / cnt + lin_b
__global__ void final_dot(const float* __restrict__ pool, const float* __restrict__ cnt,
                          const float* __restrict__ lw, const float* __restrict__ lb,
                          float* __restrict__ out) {
  int g    = (blockIdx.x * blockDim.x + threadIdx.x) >> 5;
  int lane = threadIdx.x & 31;
  if (g >= NG) return;
  const float4 p = *(const float4*)(pool + (size_t)g * DIM + lane * 4);
  const float4 w = *(const float4*)(lw + lane * 4);
  float s = p.x * w.x + p.y * w.y + p.z * w.z + p.w * w.w;
#pragma unroll
  for (int m = 16; m > 0; m >>= 1) s += __shfl_xor(s, m, 32);
  if (lane == 0) out[g] = s / fmaxf(cnt[g], 1.0f) + lb[0];
}

// ---------------------------------------------------------------- driver
extern "C" void kernel_launch(void* const* d_in, const int* in_sizes, int n_in,
                              void* d_out, int out_size, void* d_ws, size_t ws_size,
                              hipStream_t stream) {
  const float* x     = (const float*)d_in[0];
  const int*   ei    = (const int*)d_in[1];   // [2, NE] flat: row then col
  const int*   batch = (const int*)d_in[2];
  const float* Ws    = (const float*)d_in[3]; // [NL, D, D]
  const float* bs    = (const float*)d_in[4]; // [NL, D]
  const float* lw    = (const float*)d_in[5]; // [1, D]
  const float* lb    = (const float*)d_in[6]; // [1]
  float* out = (float*)d_out;

  const int* rowp = ei;
  const int* colp = ei + NE;

  // workspace carve-up (256B aligned)
  char* ws = (char*)d_ws;
  size_t off = 0;
  auto carve = [&](size_t bytes) -> void* {
    void* p = ws + off;
    off += (bytes + 255) & ~(size_t)255;
    return p;
  };
  float*  dis  = (float*) carve((size_t)NN * 4);           // deg -> deg^{-1/2}
  float*  h    = (float*) carve((size_t)NN * DIM * 4);
  float*  agg  = (float*) carve((size_t)NN * DIM * 4);
  __bf16* xb   = (__bf16*)carve((size_t)NN * DIM * 2);
  __bf16* wb   = (__bf16*)carve((size_t)NL * DIM * DIM * 2);
  float*  pool = (float*) carve((size_t)NG * DIM * 4);
  float*  cnt  = (float*) carve((size_t)NG * 4);

  const int T = 256;
  auto nb = [](int n, int t) { return (n + t - 1) / t; };
  const int ND = NN * DIM;

  // degree -> deg^{-1/2}
  fill_f32<<<nb(NN, T), T, 0, stream>>>(dis, 1.0f, NN);               // self loop
  deg_scatter<<<nb(NE, T), T, 0, stream>>>(colp, dis, NE);
  rsqrt_inplace<<<nb(NN, T), T, 0, stream>>>(dis, NN);

  // bf16 operands
  cvt_f32_bf16<<<nb(NL * DIM * DIM, T), T, 0, stream>>>(Ws, wb, NL * DIM * DIM);
  cvt_f32_bf16<<<nb(ND, T), T, 0, stream>>>(x, xb, ND);

  // pooling accumulators
  fill_f32<<<nb(NG * DIM, T), T, 0, stream>>>(pool, 0.0f, NG * DIM);
  fill_f32<<<nb(NG, T), T, 0, stream>>>(cnt, 0.0f, NG);

  const int gemm_blocks = nb(NN, 128);       // 391
  const int edge_blocks = nb(NE, T / 32);    // 8 edges (waves) per block

  for (int l = 0; l < NL; ++l) {
    gemm_xwT_wmma<<<gemm_blocks, T, 0, stream>>>(xb, wb + (size_t)l * DIM * DIM, h, NN);
    fill_f32<<<nb(ND, T), T, 0, stream>>>(agg, 0.0f, ND);
    edge_scatter<<<edge_blocks, T, 0, stream>>>(rowp, colp, dis, h, agg, NE);
    layer_finalize<<<nb(ND, T), T, 0, stream>>>(agg, h, dis, bs + (size_t)l * DIM, xb, ND);
  }

  pool_scatter<<<nb(ND, T), T, 0, stream>>>(agg, batch, pool, cnt, ND);
  final_dot<<<nb(NG * 32, T), T, 0, stream>>>(pool, cnt, lw, lb, out);
}
